// SCIEnhancementNet_66082366816704
// MI455X (gfx1250) — matrix operand order repaired and checked by reading, not compiled, not run on hardware
//
#include <hip/hip_runtime.h>

// ---------------------------------------------------------------------------
// SCI enhancement net (Mamba blocks) for MI455X / gfx1250, wave32 + WMMA.
// B=1, C=64, H=W=80, L=6400, d_inner=128, d_state=16, dt_rank=4.
// ---------------------------------------------------------------------------

typedef __attribute__((ext_vector_type(16))) _Float16 v16h;
typedef __attribute__((ext_vector_type(8)))  float    v8f;

static const int kH = 80;
static const int kW = 80;
static const int kL = 6400;   // H*W

// direction index maps (all involutions; same map for gather and scatter)
__device__ __forceinline__ int dir_map(int dir, int l) {
  switch (dir) {
    case 0:  return l;
    case 1:  return kL - 1 - l;
    case 2:  return (l % kH) * kW + (l / kH);
    default: { int lp = kL - 1 - l; return (lp % kH) * kW + (lp / kH); }
  }
}

// ---------------------------------------------------------------------------
// Generic 3x3 SAME conv, NCHW, direct.  act: 0=none 1=relu 2=sigmoid
// 3 = residual-add(res) + clip[0,1]
// ---------------------------------------------------------------------------
__global__ void conv3x3_kernel(const float* __restrict__ in,
                               const float* __restrict__ w,
                               const float* __restrict__ bias,
                               float* __restrict__ out,
                               const float* __restrict__ res,
                               int Cin, int Cout, int act) {
  int idx = blockIdx.x * blockDim.x + threadIdx.x;
  if (idx >= Cout * kH * kW) return;
  int x  = idx % kW;
  int y  = (idx / kW) % kH;
  int co = idx / (kH * kW);
  float s = bias[co];
  for (int ci = 0; ci < Cin; ++ci) {
    const float* ip = in + ci * kH * kW;
    const float* wp = w + (co * Cin + ci) * 9;
#pragma unroll
    for (int ky = 0; ky < 3; ++ky) {
      int yy = y + ky - 1;
      if (yy < 0 || yy >= kH) continue;
#pragma unroll
      for (int kx = 0; kx < 3; ++kx) {
        int xx = x + kx - 1;
        if (xx < 0 || xx >= kW) continue;
        s += ip[yy * kW + xx] * wp[ky * 3 + kx];
      }
    }
  }
  if (act == 1)      s = fmaxf(s, 0.f);
  else if (act == 2) s = 1.f / (1.f + __expf(-s));
  else if (act == 3) { s += res[idx]; s = fminf(fmaxf(s, 0.f), 1.f); }
  out[idx] = s;
}

// ---------------------------------------------------------------------------
// LayerNorm over C=64 : CHW image -> (L,64) row-major tokens
// ---------------------------------------------------------------------------
__global__ void layernorm_kernel(const float* __restrict__ img,
                                 const float* __restrict__ w,
                                 const float* __restrict__ b,
                                 float* __restrict__ xn) {
  int p = blockIdx.x * blockDim.x + threadIdx.x;
  if (p >= kL) return;
  float mu = 0.f;
  for (int c = 0; c < 64; ++c) mu += img[c * kL + p];
  mu *= (1.f / 64.f);
  float var = 0.f;
  for (int c = 0; c < 64; ++c) { float d = img[c * kL + p] - mu; var += d * d; }
  var *= (1.f / 64.f);
  float inv = rsqrtf(var + 1e-5f);
  for (int c = 0; c < 64; ++c)
    xn[p * 64 + c] = (img[c * kL + p] - mu) * inv * w[c] + b[c];
}

// gather the 4 direction-permuted sequences: xd[dir] (L,64)
__global__ void gather_kernel(const float* __restrict__ xn, float* __restrict__ xd) {
  int idx = blockIdx.x * blockDim.x + threadIdx.x;
  if (idx >= 4 * kL * 64) return;
  int dir = idx / (kL * 64);
  int r   = idx - dir * kL * 64;
  int l   = r >> 6, c = r & 63;
  xd[idx] = xn[dir_map(dir, l) * 64 + c];
}

// ---------------------------------------------------------------------------
// WMMA GEMM:  C[M=6400,N] = A[M,K] @ W[N,K]^T (+bias)  (f32 in/out, f16 math)
//
// Block = 256 threads (8 waves).  Whole weight matrix staged f32->f16 into
// LDS once per block (zero-padded to nGroups*64 rows) -> B-fragment reads are
// unconditional 32B LDS loads (2x ds_load_b128), no exec-mask branching.
// Each wave computes a 16(M) x 64(N) strip: one A fragment feeds 4 WMMAs per
// k-step (4 independent accumulators).  K and N are template parameters so
// the k-loop fully unrolls and the store guard / bias / scatter branches
// fold at compile time.
//
// blockIdx.y = direction batch.  MAPROWS: output row = dir_map(dir,row) with
// column offset dir*colPerDir (fuses the direction scatter into the store).
// ---------------------------------------------------------------------------
template <int K, int N, bool HAS_BIAS, bool MAPROWS>
__global__ void gemm_wmma_kernel(const float* __restrict__ A,
                                 const float* __restrict__ Wt,
                                 const float* __restrict__ bias,
                                 float* __restrict__ C,
                                 int ldc, int strideA, int strideC,
                                 int colPerDir) {
  constexpr int nTiles  = (N + 15) / 16;
  constexpr int nGroups = (nTiles + 3) / 4;
  constexpr int Npad    = nGroups * 64;
  constexpr int mTiles  = kL / 16;
  static_assert(Npad * K <= 16384, "LDS weight image too large");
  __shared__ _Float16 sW[Npad * K];

  int dir = blockIdx.y;
  A += dir * strideA;
  C += dir * strideC;
  int colOff = dir * colPerDir;
  int tid = threadIdx.x;

  // cooperative stage: weights -> f16 LDS, zero-pad rows [N, Npad)
  for (int idx = tid; idx < Npad * K; idx += 256) {
    int n = idx / K;
    int k = idx - n * K;
    sW[idx] = (n < N) ? (_Float16)Wt[n * K + k] : (_Float16)0.f;
  }
  __syncthreads();

  int wave = blockIdx.x * 8 + (tid >> 5);
  int lane = tid & 31;
  if (wave >= mTiles * nGroups) return;     // wave-uniform exit
  int mt   = wave % mTiles;
  int ng   = wave / mTiles;
  int half = lane >> 4;
  int l15  = lane & 15;
  int m    = mt * 16 + l15;                 // A row held by this lane

  v8f acc[4] = {};
#pragma unroll
  for (int k0 = 0; k0 < K; k0 += 32) {
    // A fragment: per lane one contiguous 16-float run -> 4x global_load_b128
    const float* ap = A + m * K + k0 + half * 16;
    float4 q0 = *(const float4*)(ap + 0);
    float4 q1 = *(const float4*)(ap + 4);
    float4 q2 = *(const float4*)(ap + 8);
    float4 q3 = *(const float4*)(ap + 12);
    v16h af;
    af[0]  = (_Float16)q0.x; af[1]  = (_Float16)q0.y;
    af[2]  = (_Float16)q0.z; af[3]  = (_Float16)q0.w;
    af[4]  = (_Float16)q1.x; af[5]  = (_Float16)q1.y;
    af[6]  = (_Float16)q1.z; af[7]  = (_Float16)q1.w;
    af[8]  = (_Float16)q2.x; af[9]  = (_Float16)q2.y;
    af[10] = (_Float16)q2.z; af[11] = (_Float16)q2.w;
    af[12] = (_Float16)q3.x; af[13] = (_Float16)q3.y;
    af[14] = (_Float16)q3.z; af[15] = (_Float16)q3.w;

#pragma unroll
    for (int j = 0; j < 4; ++j) {
      int nn = ng * 64 + j * 16 + l15;      // B column held by this lane
      // 16 contiguous f16 -> one 32B LDS read (2x ds_load_b128)
      v16h bf = *(const v16h*)(sW + nn * K + k0 + half * 16);
      acc[j] = __builtin_amdgcn_wmma_f32_16x16x32_f16(
          /*neg_a=*/false, af, /*neg_b=*/false, bf,
          /*c_mod=*/(short)0, acc[j], /*reuse_a=*/false, /*reuse_b=*/false);
    }
  }

#pragma unroll
  for (int j = 0; j < 4; ++j) {
    int n = ng * 64 + j * 16 + l15;
    if (N % 16 == 0 || n < N) {             // folds away when N % 16 == 0
      float bv = HAS_BIAS ? bias[n] : 0.f;
#pragma unroll
      for (int v = 0; v < 8; ++v) {
        int row  = mt * 16 + half * 8 + v;  // C/D layout: M = vgpr + 8*laneHalf
        int rowO = MAPROWS ? dir_map(dir, row) : row;
        C[rowO * ldc + colOff + n] = acc[j][v] + bv;
      }
    }
  }
}

// ---------------------------------------------------------------------------
// Depthwise causal conv1d (k=4) over L + SiLU.  xin = cols [0,128) of xz.
// ---------------------------------------------------------------------------
__global__ void conv1d_silu_kernel(const float* __restrict__ xz,
                                   const float* __restrict__ cw,
                                   const float* __restrict__ cb,
                                   float* __restrict__ xc) {
  int idx = blockIdx.x * blockDim.x + threadIdx.x;
  if (idx >= 4 * kL * 128) return;
  int dir = idx / (kL * 128);
  int r   = idx - dir * kL * 128;
  int l = r >> 7, d = r & 127;
  const float* xzp = xz + dir * kL * 256;
  float s = cb[d];
#pragma unroll
  for (int k = 0; k < 4; ++k) {
    int j = l - 3 + k;
    if (j >= 0) s += xzp[j * 256 + d] * cw[d * 4 + k];
  }
  xc[idx] = s / (1.f + __expf(-s));          // silu
}

// delta = softplus(dt @ dt_w^T + dt_b)  with dt = dbl cols [0,4)
__global__ void dt_kernel(const float* __restrict__ dbl,
                          const float* __restrict__ dtw,
                          const float* __restrict__ dtb,
                          float* __restrict__ delta) {
  int idx = blockIdx.x * blockDim.x + threadIdx.x;
  if (idx >= 4 * kL * 128) return;
  int dir = idx / (kL * 128);
  int r   = idx - dir * kL * 128;
  int l = r >> 7, d = r & 127;
  const float* row = dbl + dir * kL * 36 + l * 36;
  float s = dtb[d];
#pragma unroll
  for (int k = 0; k < 4; ++k) s += row[k] * dtw[d * 4 + k];
  delta[idx] = (s > 20.f) ? s : log1pf(__expf(s));
}

__global__ void prepA_kernel(const float* __restrict__ A_log, float* __restrict__ Aneg) {
  int idx = blockIdx.x * blockDim.x + threadIdx.x;
  if (idx < 128 * 16) Aneg[idx] = -__expf(A_log[idx]);
}

// ---------------------------------------------------------------------------
// Selective scan, one block per direction (4 blocks run concurrently).
// 256 threads; thread owns (d = tid>>1, n = (tid&1)*8 .. +7) states.
// Pair-reduce over n with wave32 shuffle.  Output gated by silu(z).
// ---------------------------------------------------------------------------
__global__ void scan_kernel(const float* __restrict__ xc,
                            const float* __restrict__ delta,
                            const float* __restrict__ dbl,
                            const float* __restrict__ Aneg,
                            const float* __restrict__ Dvec,
                            const float* __restrict__ xz,
                            float* __restrict__ yg) {
  int dir = blockIdx.x;
  const float* u  = xc    + dir * kL * 128;
  const float* dl = delta + dir * kL * 128;
  const float* bc = dbl   + dir * kL * 36;
  const float* zz = xz    + dir * kL * 256;
  float*       yo = yg    + dir * kL * 128;

  int tid = threadIdx.x;
  int d  = tid >> 1;
  int n0 = (tid & 1) * 8;
  float h[8], a[8];
#pragma unroll
  for (int j = 0; j < 8; ++j) { h[j] = 0.f; a[j] = Aneg[d * 16 + n0 + j]; }
  float Dd = Dvec[d];

  for (int t = 0; t < kL; ++t) {
    float dt_ = dl[t * 128 + d];
    float ut  = u [t * 128 + d];
    const float* row = bc + t * 36;
    float acc = 0.f;
#pragma unroll
    for (int j = 0; j < 8; ++j) {
      float Bj = row[4 + n0 + j];
      float Cj = row[20 + n0 + j];
      float dA = __expf(dt_ * a[j]);
      h[j] = dA * h[j] + dt_ * Bj * ut;
      acc += h[j] * Cj;
    }
    acc += __shfl_xor(acc, 1, 32);           // combine the two n-halves (wave32)
    if ((tid & 1) == 0) {
      float zv = zz[t * 256 + 128 + d];
      float gate = zv / (1.f + __expf(-zv)); // silu(z)
      yo[t * 128 + d] = (acc + ut * Dd) * gate;
    }
    if (t + 1 < kL) {                        // gfx1250 global_prefetch_b8
      __builtin_prefetch(&dl[(t + 1) * 128 + d], 0, 0);
      __builtin_prefetch(&u [(t + 1) * 128 + d], 0, 0);
    }
  }
}

// feat(CHW) += fused(L,64)
__global__ void resadd_kernel(float* __restrict__ feat, const float* __restrict__ fo) {
  int idx = blockIdx.x * blockDim.x + threadIdx.x;
  if (idx >= 64 * kL) return;
  int c = idx / kL, p = idx - c * kL;
  feat[idx] += fo[p * 64 + c];
}

// feat = feat * (1 + g) + b    (all CHW, same layout)
__global__ void sft_kernel(float* __restrict__ feat,
                           const float* __restrict__ g,
                           const float* __restrict__ b) {
  int idx = blockIdx.x * blockDim.x + threadIdx.x;
  if (idx >= 64 * kL) return;
  feat[idx] = feat[idx] * (1.f + g[idx]) + b[idx];
}

// ---------------------------------------------------------------------------
// Host launcher
// ---------------------------------------------------------------------------
extern "C" void kernel_launch(void* const* d_in, const int* in_sizes, int n_in,
                              void* d_out, int out_size, void* d_ws, size_t ws_size,
                              hipStream_t stream) {
  const float* x      = (const float*)d_in[0];
  const float* eb_w1  = (const float*)d_in[1];
  const float* eb_b1  = (const float*)d_in[2];
  const float* eb_w2  = (const float*)d_in[3];
  const float* eb_b2  = (const float*)d_in[4];
  const float* eb_w3  = (const float*)d_in[5];
  const float* eb_b3  = (const float*)d_in[6];
  const float* ci_w   = (const float*)d_in[7];
  const float* ci_b   = (const float*)d_in[8];
  const float* ln_w   = (const float*)d_in[9];
  const float* ln_b   = (const float*)d_in[10];
  const float* in_w   = (const float*)d_in[11];
  const float* conv_w = (const float*)d_in[12];
  const float* conv_b = (const float*)d_in[13];
  const float* xp_w   = (const float*)d_in[14];
  const float* dt_w   = (const float*)d_in[15];
  const float* dt_b   = (const float*)d_in[16];
  const float* A_log  = (const float*)d_in[17];
  const float* Dp     = (const float*)d_in[18];
  const float* out_w  = (const float*)d_in[19];
  const float* fuse_w = (const float*)d_in[20];
  const float* fuse_b = (const float*)d_in[21];
  const float* sft_gw = (const float*)d_in[22];
  const float* sft_gb = (const float*)d_in[23];
  const float* sft_bw = (const float*)d_in[24];
  const float* sft_bb = (const float*)d_in[25];
  const float* co_w   = (const float*)d_in[26];
  const float* co_b   = (const float*)d_in[27];

  float* outp = (float*)d_out;           // (1,3,80,80) = 19200
  float* mask = outp + 3 * kL;           // (1,1,80,80) = 6400 (2nd tuple output)

  // workspace layout (floats)
  float* wsf   = (float*)d_ws;
  float* f1    = wsf;                    // 32*L
  float* f2    = f1    + 32 * kL;        // 32*L
  float* feat  = f2    + 32 * kL;        // 64*L
  float* xn    = feat  + 64 * kL;        // L*64
  float* xd    = xn    + 64 * kL;        // 4*L*64
  float* xz    = xd    + 4 * kL * 64;    // 4*L*256
  float* xc    = xz    + 4 * kL * 256;   // 4*L*128
  float* dbl   = xc    + 4 * kL * 128;   // 4*L*36
  float* delta = dbl   + 4 * kL * 36;    // 4*L*128
  float* Aneg  = delta + 4 * kL * 128;   // 2048
  float* yg    = Aneg  + 2048;           // 4*L*128
  float* fin   = yg    + 4 * kL * 128;   // L*256
  float* fout  = fin   + kL * 256;       // L*64
  float* gimg  = fout  + kL * 64;        // 64*L
  float* bimg  = gimg  + 64 * kL;        // 64*L

  auto cdiv = [](int a, int b) { return (a + b - 1) / b; };
  const int T = 256;

  // ----- shallow encoder + mask + input conv -----
  conv3x3_kernel<<<cdiv(32 * kL, T), T, 0, stream>>>(x,  eb_w1, eb_b1, f1, nullptr, 3, 32, 1);
  conv3x3_kernel<<<cdiv(32 * kL, T), T, 0, stream>>>(f1, eb_w2, eb_b2, f2, nullptr, 32, 32, 1);
  conv3x3_kernel<<<cdiv(kL, T),      T, 0, stream>>>(f2, eb_w3, eb_b3, mask, nullptr, 32, 1, 2);
  conv3x3_kernel<<<cdiv(64 * kL, T), T, 0, stream>>>(x,  ci_w,  ci_b,  feat, nullptr, 3, 64, 1);

  // ----- one mamba block (all 4 directions batched on grid.y) -----
  auto mblock = [&](int i) {
    layernorm_kernel<<<cdiv(kL, T), T, 0, stream>>>(feat, ln_w + i * 64, ln_b + i * 64, xn);
    gather_kernel<<<cdiv(4 * kL * 64, T), T, 0, stream>>>(xn, xd);

    // in-proj: (L,64) @ (256,64)^T -> xz (L,256): 400 mTiles x 4 nGroups
    {
      dim3 g(cdiv(400 * 4, 8), 4);
      gemm_wmma_kernel<64, 256, false, false><<<g, 256, 0, stream>>>(
          xd, in_w + i * 256 * 64, nullptr, xz, 256, kL * 64, kL * 256, 0);
    }
    conv1d_silu_kernel<<<cdiv(4 * kL * 128, T), T, 0, stream>>>(
        xz, conv_w + i * 512, conv_b + i * 128, xc);

    // x-proj: (L,128) @ (36,128)^T -> dbl (L,36): 400 mTiles x 1 nGroup
    {
      dim3 g(cdiv(400, 8), 4);
      gemm_wmma_kernel<128, 36, false, false><<<g, 256, 0, stream>>>(
          xc, xp_w + i * 36 * 128, nullptr, dbl, 36, kL * 128, kL * 36, 0);
    }
    dt_kernel<<<cdiv(4 * kL * 128, T), T, 0, stream>>>(
        dbl, dt_w + i * 512, dt_b + i * 128, delta);
    prepA_kernel<<<cdiv(2048, T), T, 0, stream>>>(A_log + i * 2048, Aneg);

    // 4 directions scanned concurrently
    scan_kernel<<<4, 256, 0, stream>>>(xc, delta, dbl, Aneg, Dp + i * 128, xz, yg);

    // out-proj with fused direction scatter into fin (L,256)
    {
      dim3 g(cdiv(400, 8), 4);
      gemm_wmma_kernel<128, 64, false, true><<<g, 256, 0, stream>>>(
          yg, out_w + i * 64 * 128, nullptr, fin, 256, kL * 128, 0, 64);
    }
    // fuse: (L,256) @ (64,256)^T + bias -> fout (L,64)
    {
      dim3 g(cdiv(400, 8), 1);
      gemm_wmma_kernel<256, 64, true, false><<<g, 256, 0, stream>>>(
          fin, fuse_w + i * 64 * 256, fuse_b + i * 64, fout, 64, 0, 0, 0);
    }
    resadd_kernel<<<cdiv(64 * kL, T), T, 0, stream>>>(feat, fout);
  };

  auto sft = [&](int s) {
    conv3x3_kernel<<<cdiv(64 * kL, T), T, 0, stream>>>(mask, sft_gw + s * 576, sft_gb + s * 64,
                                                       gimg, nullptr, 1, 64, 0);
    conv3x3_kernel<<<cdiv(64 * kL, T), T, 0, stream>>>(mask, sft_bw + s * 576, sft_bb + s * 64,
                                                       bimg, nullptr, 1, 64, 0);
    sft_kernel<<<cdiv(64 * kL, T), T, 0, stream>>>(feat, gimg, bimg);
  };

  mblock(0); mblock(1); sft(0);
  mblock(2); mblock(3); sft(1);

  // final conv + residual + clip -> out
  conv3x3_kernel<<<cdiv(3 * kL, T), T, 0, stream>>>(feat, co_w, co_b, outp, x, 64, 3, 3);
}
